// CoupletsAttentionModel_86251533238989
// MI455X (gfx1250) — matrix operand order repaired and checked by reading, not compiled, not run on hardware
//
#include <hip/hip_runtime.h>
#include <hip/hip_bf16.h>
#include <stdint.h>

#define NB    8
#define S_ENC 2048
#define S_DEC 2048
#define DIM   512

typedef __attribute__((ext_vector_type(16))) __bf16         v16bf;
typedef __attribute__((ext_vector_type(8)))  float          v8f;
typedef __attribute__((ext_vector_type(8)))  unsigned short ushort8;
typedef __attribute__((ext_vector_type(4)))  unsigned short ushort4v;

__device__ __forceinline__ unsigned short f2bf(float f) {
    union { float f; unsigned int u; } c; c.f = f;
    unsigned int r = c.u + 0x7FFFu + ((c.u >> 16) & 1u);
    return (unsigned short)(r >> 16);
}

__device__ __forceinline__ v8f wmma_bf16(v16bf a, v16bf b, v8f c) {
    // v_wmma_f32_16x16x32_bf16: D = A(16x32) * B(32x16) + C(16x16 f32)
    return __builtin_amdgcn_wmma_f32_16x16x32_bf16(
        /*neg_a=*/false, a, /*neg_b=*/false, b,
        /*c_mod=*/(short)0, c, /*reuse_a=*/false, /*reuse_b=*/false);
}

union bfrag16 { v16bf v; ushort8 h[2]; };

// ---------------------------------------------------------------------------
// fp32 -> bf16 elementwise convert (4 elems/thread, b128-in / b64-out)
// ---------------------------------------------------------------------------
__global__ void convert_bf16_kernel(const float* __restrict__ src,
                                    unsigned short* __restrict__ dst, long n) {
    long idx = ((long)blockIdx.x * blockDim.x + threadIdx.x) * 4;
    if (idx + 3 < n) {
        float4 v = *(const float4*)(src + idx);
        ushort4v o;
        o[0] = f2bf(v.x); o[1] = f2bf(v.y); o[2] = f2bf(v.z); o[3] = f2bf(v.w);
        *(ushort4v*)(dst + idx) = o;
    }
}

// ---------------------------------------------------------------------------
// enc fp32 [b][s][d] -> bf16 encT [b][d][s], 32x32 LDS-tiled transpose
// ---------------------------------------------------------------------------
__global__ void transpose_bf16_kernel(const float* __restrict__ enc,
                                      unsigned short* __restrict__ encT) {
    __shared__ unsigned short tile[32][33];
    const int b  = blockIdx.z;
    const int s0 = blockIdx.y * 32;
    const int d0 = blockIdx.x * 32;
    const int tx = threadIdx.x;   // 0..31
    const int ty = threadIdx.y;   // 0..7
    #pragma unroll
    for (int r = ty; r < 32; r += 8)
        tile[r][tx] = f2bf(enc[((size_t)b * S_ENC + s0 + r) * DIM + d0 + tx]);
    __syncthreads();
    #pragma unroll
    for (int r = ty; r < 32; r += 8)
        encT[((size_t)b * DIM + d0 + r) * S_ENC + s0 + tx] = tile[tx][r];
}

// ---------------------------------------------------------------------------
// Flash attention: block = 8 waves (256 thr), owns (batch, 32 q rows).
// Each wave: 2 q-subtiles of 16 rows, a 64-wide D slice of the output.
// k streamed in steps of 32. 16 WMMAs / wave / k-step.
// ---------------------------------------------------------------------------
__global__ __launch_bounds__(256)
void fa_kernel(const unsigned short* __restrict__ decB,   // bf16 [b][s_dec][d]
               const unsigned short* __restrict__ encB,   // bf16 [b][s_enc][d]
               const unsigned short* __restrict__ encT,   // bf16 [b][d][s_enc]
               float* __restrict__ out)                   // f32  [b][s_dec][d]
{
    __shared__ float          sS[8][32 * 32];                // per-wave partial scores
    __shared__ alignas(16) unsigned short sP[32 * 32];       // bf16 probs
    __shared__ float          sM[32], sL[32], sAlpha[32];

    const int b     = blockIdx.y;
    const int qbase = blockIdx.x * 32;
    const int tid   = threadIdx.x;
    const int wave  = tid >> 5;
    const int lane  = tid & 31;
    const int hh    = lane >> 4;       // lane half (K/M split per ISA layouts)
    const int lmod  = lane & 15;
    const int dw    = wave * 64;       // this wave's output D slice

    if (tid < 32) { sM[tid] = -3.0e38f; sL[tid] = 0.0f; }

    // ---- Q A-fragments: 2 subtiles x 2 d-chunks (loop invariant)
    bfrag16 qa[2][2];
    #pragma unroll
    for (int t = 0; t < 2; ++t) {
        const unsigned short* p = decB +
            ((size_t)b * S_DEC + qbase + t * 16 + lmod) * DIM + dw + hh * 8;
        qa[t][0].h[0] = *(const ushort8*)(p);
        qa[t][0].h[1] = *(const ushort8*)(p + 16);
        qa[t][1].h[0] = *(const ushort8*)(p + 32);
        qa[t][1].h[1] = *(const ushort8*)(p + 48);
    }

    v8f o[2][4] = {{v8f{}, v8f{}, v8f{}, v8f{}}, {v8f{}, v8f{}, v8f{}, v8f{}}};

    const unsigned short* tb = encT + (size_t)b * DIM * S_ENC;

    for (int kb = 0; kb < S_ENC; kb += 32) {
        // ---- load ALL score B-fragments first (lets scheduler clause loads)
        // bK[c][nh]: c = d-chunk (dw, dw+32), nh = enc-col half (kb, kb+16)
        const unsigned short* p0 = encB + ((size_t)b * S_ENC + kb + lmod) * DIM + dw + hh * 8;
        const unsigned short* p1 = p0 + 16 * DIM;
        bfrag16 bK[2][2];
        bK[0][0].h[0] = *(const ushort8*)(p0);
        bK[0][0].h[1] = *(const ushort8*)(p0 + 16);
        bK[0][1].h[0] = *(const ushort8*)(p1);
        bK[0][1].h[1] = *(const ushort8*)(p1 + 16);
        bK[1][0].h[0] = *(const ushort8*)(p0 + 32);
        bK[1][0].h[1] = *(const ushort8*)(p0 + 48);
        bK[1][1].h[0] = *(const ushort8*)(p1 + 32);
        bK[1][1].h[1] = *(const ushort8*)(p1 + 48);

        if (kb + 32 < S_ENC) {                    // prefetch this k-tile's rows at d+32 into GL2/L0
            __builtin_prefetch(p0 + 32, 0, 3);
            __builtin_prefetch(p1 + 32, 0, 3);
        }

        // ---- partial S (2 subtiles x 16x32) over this wave's 64-wide D slice
        v8f s[2][2] = {{v8f{}, v8f{}}, {v8f{}, v8f{}}};
        #pragma unroll
        for (int t = 0; t < 2; ++t) {
            #pragma unroll
            for (int nh = 0; nh < 2; ++nh) {
                s[t][nh] = wmma_bf16(qa[t][0].v, bK[0][nh].v, s[t][nh]);
                s[t][nh] = wmma_bf16(qa[t][1].v, bK[1][nh].v, s[t][nh]);
            }
        }

        // ---- park partials in LDS (C/D layout: elem r -> row r+8*hh, col lmod)
        {
            float* srow = &sS[wave][0];
            #pragma unroll
            for (int t = 0; t < 2; ++t) {
                #pragma unroll
                for (int r = 0; r < 8; ++r) {
                    srow[(t * 16 + r + 8 * hh) * 32 + lmod]      = s[t][0][r];
                    srow[(t * 16 + r + 8 * hh) * 32 + 16 + lmod] = s[t][1][r];
                }
            }
        }
        __syncthreads();

        // ---- cooperative online softmax: thread -> (row = tid>>3, 4 cols)
        {
            const int row = tid >> 3;      // 0..31
            const int j   = tid & 7;
            const int c0  = j * 4;
            float4 a = {0.f, 0.f, 0.f, 0.f};
            #pragma unroll
            for (int w = 0; w < 8; ++w) {
                float4 v = *(const float4*)(&sS[w][row * 32 + c0]);
                a.x += v.x; a.y += v.y; a.z += v.z; a.w += v.w;
            }
            float mOld = sM[row];
            float mx = fmaxf(fmaxf(a.x, a.y), fmaxf(a.z, a.w));
            #pragma unroll
            for (int m = 4; m >= 1; m >>= 1) mx = fmaxf(mx, __shfl_xor(mx, m, 8));
            float mNew  = fmaxf(mOld, mx);
            float alpha = __expf(mOld - mNew);
            float px = __expf(a.x - mNew);
            float py = __expf(a.y - mNew);
            float pz = __expf(a.z - mNew);
            float pw = __expf(a.w - mNew);
            float ps = (px + py) + (pz + pw);
            #pragma unroll
            for (int m = 4; m >= 1; m >>= 1) ps += __shfl_xor(ps, m, 8);
            ushort4v pb;
            pb[0] = f2bf(px); pb[1] = f2bf(py); pb[2] = f2bf(pz); pb[3] = f2bf(pw);
            *(ushort4v*)(&sP[row * 32 + c0]) = pb;
            if (j == 0) {
                sL[row]     = sL[row] * alpha + ps;
                sM[row]     = mNew;
                sAlpha[row] = alpha;
            }
        }
        __syncthreads();

        // ---- load ALL V B-fragments + both P A-fragments, then rescale + WMMA
        const unsigned short* vrow =
            tb + (size_t)(dw + lmod) * S_ENC + kb + hh * 8;   // lmod = N (d col), elems = K (enc)
        bfrag16 vf[4];
        #pragma unroll
        for (int nt = 0; nt < 4; ++nt) {
            vf[nt].h[0] = *(const ushort8*)(vrow + (size_t)nt * 16 * S_ENC);
            vf[nt].h[1] = *(const ushort8*)(vrow + (size_t)nt * 16 * S_ENC + 16);
        }
        if (kb + 32 < S_ENC)
            __builtin_prefetch(vrow + 32 * DIM, 0, 3);

        bfrag16 pf[2];
        #pragma unroll
        for (int t = 0; t < 2; ++t) {
            const unsigned short* pp = &sP[(t * 16 + lmod) * 32 + hh * 8];
            pf[t].h[0] = *(const ushort8*)(pp);          // K = hh*8..+7
            pf[t].h[1] = *(const ushort8*)(pp + 16);     // K = 16+hh*8..+7
        }

        #pragma unroll
        for (int t = 0; t < 2; ++t) {
            #pragma unroll
            for (int r = 0; r < 8; ++r) {
                float a = sAlpha[t * 16 + r + 8 * hh];
                o[t][0][r] *= a; o[t][1][r] *= a; o[t][2][r] *= a; o[t][3][r] *= a;
            }
        }
        #pragma unroll
        for (int t = 0; t < 2; ++t)
            #pragma unroll
            for (int nt = 0; nt < 4; ++nt)
                o[t][nt] = wmma_bf16(pf[t].v, vf[nt].v, o[t][nt]);
        // next iteration's __syncthreads() (after sS stores) fences sP/sAlpha reuse
    }

    // ---- normalize and store (after last barrier, sL is final)
    #pragma unroll
    for (int t = 0; t < 2; ++t) {
        #pragma unroll
        for (int r = 0; r < 8; ++r) {
            const int row = t * 16 + r + 8 * hh;
            const float inv = 1.0f / sL[row];
            float* orow = out + ((size_t)b * S_DEC + qbase + row) * DIM + dw + lmod;
            orow[0]  = o[t][0][r] * inv;
            orow[16] = o[t][1][r] * inv;
            orow[32] = o[t][2][r] * inv;
            orow[48] = o[t][3][r] * inv;
        }
    }
}

// ---------------------------------------------------------------------------
extern "C" void kernel_launch(void* const* d_in, const int* in_sizes, int n_in,
                              void* d_out, int out_size, void* d_ws, size_t ws_size,
                              hipStream_t stream) {
    const float* enc = (const float*)d_in[0];   // [8, 2048, 512]
    const float* dec = (const float*)d_in[1];   // [8, 2048, 512]
    float* out = (float*)d_out;                 // [8, 2048, 512]

    const size_t nDec = (size_t)NB * S_DEC * DIM;
    const size_t nEnc = (size_t)NB * S_ENC * DIM;
    unsigned short* decB = (unsigned short*)d_ws;
    unsigned short* encB = decB + nDec;
    unsigned short* encT = encB + nEnc;

    convert_bf16_kernel<<<(unsigned)(nDec / 4 / 256), 256, 0, stream>>>(dec, decB, (long)nDec);
    convert_bf16_kernel<<<(unsigned)(nEnc / 4 / 256), 256, 0, stream>>>(enc, encB, (long)nEnc);
    transpose_bf16_kernel<<<dim3(DIM / 32, S_ENC / 32, NB), dim3(32, 8), 0, stream>>>(enc, encT);

    fa_kernel<<<dim3(S_DEC / 32, NB), 256, 0, stream>>>(decB, encB, encT, out);
}
